// GNNEncoder_57071525429486
// MI455X (gfx1250) — matrix operand order repaired and checked by reading, not compiled, not run on hardware
//
#include <hip/hip_runtime.h>
#include <stdint.h>

typedef __attribute__((ext_vector_type(16))) _Float16 v16h;
typedef __attribute__((ext_vector_type(8)))  float    v8f;

#define HDIM 64

// ---------------- zero fill ----------------
__global__ void k_zero(float* __restrict__ p, long long n) {
    long long i = (long long)blockIdx.x * blockDim.x + threadIdx.x;
    long long stride = (long long)gridDim.x * blockDim.x;
    for (; i < n; i += stride) p[i] = 0.0f;
}

// ---------------- degree counts (both directions in one pass) ----------------
__global__ void k_deg(const int* __restrict__ src, const int* __restrict__ dst,
                      float* __restrict__ deg_u, float* __restrict__ deg_m, long long E) {
    long long e = (long long)blockIdx.x * blockDim.x + threadIdx.x;
    if (e >= E) return;
    atomicAdd(deg_u + src[e], 1.0f);
    atomicAdd(deg_m + dst[e], 1.0f);
}

// ---------------- fold linear-bias + BN(eval) into per-column affine ----------------
__global__ void k_affine(const float* __restrict__ lb, const float* __restrict__ g,
                         const float* __restrict__ bb, const float* __restrict__ rm,
                         const float* __restrict__ rv, float* __restrict__ s, float* __restrict__ c) {
    int i = threadIdx.x;  // 64 threads
    float sv = g[i] * rsqrtf(rv[i] + 1e-5f);
    s[i] = sv;
    c[i] = (lb[i] - rm[i]) * sv + bb[i];
}

// ---------------- edge scatter: sum[dst] += feat[src], 1 edge per 32 lanes ----------------
__global__ void k_scatter(const float* __restrict__ feat, const int* __restrict__ src,
                          const int* __restrict__ dst, float* __restrict__ sum, long long E) {
    long long t = (long long)blockIdx.x * blockDim.x + threadIdx.x;
    long long e = t >> 5;
    int li = (int)(t & 31);
    if (e >= E) return;
    int s = src[e], d = dst[e];
    const float2 v = *reinterpret_cast<const float2*>(feat + (long long)s * HDIM + li * 2);
    float* out = sum + (long long)d * HDIM + li * 2;
    atomicAdd(out, v.x);
    atomicAdd(out + 1, v.y);
}

// ---------------- fused dual-GEMM + affine + relu (WMMA f16->f32) ----------------
// out[r, :] = act( ((A1[r,:]*invdeg(r)) @ W1 + (A2 ? A2[r,:] @ W2 : 0)) * s + c )
// W row-major [64 (in/K), 64 (out/N)]. One wave per 16-row tile, 8 waves/block.
__global__ __launch_bounds__(256) void k_gemm64(
    const float* __restrict__ A1, const float* __restrict__ deg,
    const float* __restrict__ W1g,
    const float* __restrict__ A2, const float* __restrict__ W2g,
    const float* __restrict__ scl, const float* __restrict__ sft,
    float* __restrict__ out, int nrows, int relu) {
    __shared__ float w1[HDIM * HDIM];
    __shared__ float w2[HDIM * HDIM];
    const int tid = threadIdx.x;
    for (int i = tid; i < HDIM * HDIM; i += 256) {
        w1[i] = W1g[i];
        if (A2) w2[i] = W2g[i];
    }
    __syncthreads();

    const int wave = tid >> 5;
    const int lane = tid & 31;
    const int row0 = blockIdx.x * 128 + wave * 16;
    if (row0 >= nrows) return;  // nrows % 16 == 0 for this problem; wave-uniform exit

    const int hi = lane >> 4;       // 0: lanes 0-15, 1: lanes 16-31
    const int ln = lane & 15;
    const int mrow = row0 + ln;     // A-fragment row for this lane

    v8f acc[4];
#pragma unroll
    for (int j = 0; j < 4; ++j) acc[j] = (v8f)0.0f;

    const int npass = A2 ? 2 : 1;
    for (int p = 0; p < npass; ++p) {
        const float* __restrict__ A = p ? A2 : A1;
        const float* W = p ? w2 : w1;
        float rs = 1.0f;
        if (p == 0 && deg) rs = 1.0f / fmaxf(deg[mrow], 1.0f);  // scatter-mean divide, fused
        const float* __restrict__ arow = A + (long long)mrow * HDIM;
#pragma unroll
        for (int kh = 0; kh < 2; ++kh) {
            // A fragment: 16-bit A 16x32 layout (ISA 7.12.2):
            // lane half 0 holds K 0..7,16..23 ; half 1 holds K 8..15,24..31 (pairs per VGPR)
            v16h a;
#pragma unroll
            for (int v = 0; v < 8; ++v) {
                int k = 32 * kh + (v < 4 ? 2 * v : 2 * v + 8) + hi * 8;
                a[2 * v]     = (_Float16)(arow[k] * rs);
                a[2 * v + 1] = (_Float16)(arow[k + 1] * rs);
            }
#pragma unroll
            for (int j = 0; j < 4; ++j) {
                // B fragment: 16-bit B 32x16 layout: N = lane%16,
                // lane half 0 holds K 0..7,16..23 ; half 1 holds K 8..15,24..31 (pairs per VGPR)
                v16h b;
                int n = ln + 16 * j;
#pragma unroll
                for (int v = 0; v < 8; ++v) {
                    int k = 32 * kh + (v < 4 ? 2 * v : 2 * v + 8) + hi * 8;
                    b[2 * v]     = (_Float16)W[k * HDIM + n];
                    b[2 * v + 1] = (_Float16)W[(k + 1) * HDIM + n];
                }
                acc[j] = __builtin_amdgcn_wmma_f32_16x16x32_f16(
                    false, a, false, b, (short)0, acc[j], false, false);
            }
        }
    }

    // Epilogue: C/D layout — VGPR i holds M = i + 8*hi, N = lane%16 (+16j)
#pragma unroll
    for (int j = 0; j < 4; ++j) {
        int n = ln + 16 * j;
        float sj = scl ? scl[n] : 1.0f;
        float cj = sft ? sft[n] : 0.0f;
#pragma unroll
        for (int i = 0; i < 8; ++i) {
            int m = i + hi * 8;
            float v = fmaf(acc[j][i], sj, cj);
            if (relu) v = fmaxf(v, 0.0f);
            out[(long long)(row0 + m) * HDIM + n] = v;
        }
    }
}

extern "C" void kernel_launch(void* const* d_in, const int* in_sizes, int n_in,
                              void* d_out, int out_size, void* d_ws, size_t ws_size,
                              hipStream_t stream) {
    const float* x_user  = (const float*)d_in[0];
    const float* x_movie = (const float*)d_in[1];
    const int*   e_src   = (const int*)d_in[2];  // user index per edge
    const int*   e_dst   = (const int*)d_in[3];  // movie index per edge
    const float* lin_w_u = (const float*)d_in[4];
    const float* lin_b_u = (const float*)d_in[5];
    const float* lin_w_m = (const float*)d_in[6];
    const float* lin_b_m = (const float*)d_in[7];
    const float* bn_g_u = (const float*)d_in[8],  *bn_b_u = (const float*)d_in[9];
    const float* bn_rm_u = (const float*)d_in[10], *bn_rv_u = (const float*)d_in[11];
    const float* bn_g_m = (const float*)d_in[12], *bn_b_m = (const float*)d_in[13];
    const float* bn_rm_m = (const float*)d_in[14], *bn_rv_m = (const float*)d_in[15];
    const float* c1_um_wl = (const float*)d_in[16], *c1_um_bl = (const float*)d_in[17], *c1_um_wr = (const float*)d_in[18];
    const float* c1_mu_wl = (const float*)d_in[19], *c1_mu_bl = (const float*)d_in[20], *c1_mu_wr = (const float*)d_in[21];
    const float* c2_um_wl = (const float*)d_in[22], *c2_um_bl = (const float*)d_in[23], *c2_um_wr = (const float*)d_in[24];
    const float* c2_mu_wl = (const float*)d_in[25], *c2_mu_bl = (const float*)d_in[26], *c2_mu_wr = (const float*)d_in[27];

    const long long NU = in_sizes[0] / HDIM;
    const long long NM = in_sizes[1] / HDIM;
    const long long E  = in_sizes[2];

    // Workspace layout (floats)
    float* ws    = (float*)d_ws;
    float* bufA  = ws;                   // NU*64 : hu, later sum_u2
    float* bufB  = bufA + NU * HDIM;     // NM*64 : hm, later sum_m2
    float* bufC  = bufB + NM * HDIM;     // NU*64 : sum_u, then u1 (aliased)
    float* bufD  = bufC + NU * HDIM;     // NM*64 : sum_m, then m1 (aliased)
    float* deg_u = bufD + NM * HDIM;     // NU
    float* deg_m = deg_u + NU;           // NM
    float* s_u = deg_m + NM;  float* c_u = s_u + 64;
    float* s_m = c_u + 64;    float* c_m = s_m + 64;

    const int TB = 256;
    const int gz_deg  = (int)((NU + NM + TB - 1) / TB);
    const int gz_nu   = (int)((NU * HDIM + TB - 1) / TB);
    const int gz_nm   = (int)((NM * HDIM + TB - 1) / TB);
    const int g_edge  = (int)((E + TB - 1) / TB);
    const int g_scat  = (int)((E * 32 + TB - 1) / TB);
    const int g_gu    = (int)((NU + 127) / 128);
    const int g_gm    = (int)((NM + 127) / 128);

    float* out_user  = (float*)d_out;
    float* out_movie = out_user + NU * HDIM;

    // degrees (shared by both layers)
    k_zero<<<gz_deg, TB, 0, stream>>>(deg_u, NU + NM);  // deg_u,deg_m contiguous
    k_deg<<<g_edge, TB, 0, stream>>>(e_src, e_dst, deg_u, deg_m, E);

    // folded BN affine
    k_affine<<<1, 64, 0, stream>>>(lin_b_u, bn_g_u, bn_b_u, bn_rm_u, bn_rv_u, s_u, c_u);
    k_affine<<<1, 64, 0, stream>>>(lin_b_m, bn_g_m, bn_b_m, bn_rm_m, bn_rv_m, s_m, c_m);

    // hu = relu(BN(x_user @ W + b)), hm likewise
    k_gemm64<<<g_gu, 256, 0, stream>>>(x_user,  nullptr, lin_w_u, nullptr, nullptr, s_u, c_u, bufA, (int)NU, 1);
    k_gemm64<<<g_gm, 256, 0, stream>>>(x_movie, nullptr, lin_w_m, nullptr, nullptr, s_m, c_m, bufB, (int)NM, 1);

    // conv1 aggregation
    k_zero<<<gz_nu, TB, 0, stream>>>(bufC, NU * HDIM);  // sum_u
    k_zero<<<gz_nm, TB, 0, stream>>>(bufD, NM * HDIM);  // sum_m
    k_scatter<<<g_scat, TB, 0, stream>>>(bufA, e_src, e_dst, bufD, E);  // hu -> movie sums
    k_scatter<<<g_scat, TB, 0, stream>>>(bufB, e_dst, e_src, bufC, E);  // hm -> user sums

    // conv1 SAGE: out = relu(mean@wl + bl + xdst@wr)   (output aliases its own sum buffer: safe, tile-local)
    k_gemm64<<<g_gm, 256, 0, stream>>>(bufD, deg_m, c1_um_wl, bufB, c1_um_wr, nullptr, c1_um_bl, bufD, (int)NM, 1);  // m1
    k_gemm64<<<g_gu, 256, 0, stream>>>(bufC, deg_u, c1_mu_wl, bufA, c1_mu_wr, nullptr, c1_mu_bl, bufC, (int)NU, 1);  // u1

    // conv2 aggregation (reuse hu/hm buffers as sums)
    k_zero<<<gz_nu, TB, 0, stream>>>(bufA, NU * HDIM);  // sum_u2
    k_zero<<<gz_nm, TB, 0, stream>>>(bufB, NM * HDIM);  // sum_m2
    k_scatter<<<g_scat, TB, 0, stream>>>(bufC, e_src, e_dst, bufB, E);  // u1 -> movie sums
    k_scatter<<<g_scat, TB, 0, stream>>>(bufD, e_dst, e_src, bufA, E);  // m1 -> user sums

    // conv2 SAGE (no relu), straight into d_out (out_user first, then out_movie)
    k_gemm64<<<g_gm, 256, 0, stream>>>(bufB, deg_m, c2_um_wl, bufD, c2_um_wr, nullptr, c2_um_bl, out_movie, (int)NM, 0);
    k_gemm64<<<g_gu, 256, 0, stream>>>(bufA, deg_u, c2_mu_wl, bufC, c2_mu_wr, nullptr, c2_mu_bl, out_user, (int)NU, 0);
}
